// NonlinearReducedBasisSurrogate_72748156059688
// MI455X (gfx1250) — matrix operand order repaired and checked by reading, not compiled, not run on hardware
//
#include <hip/hip_runtime.h>
#include <cstdint>
#include <cstddef>

// ---------------- problem constants (from reference) ----------------
#define N_IN   131072
#define L0     512
#define L1     256
#define NLAT   64
#define WIN    512          // 2*MU
#define TM     256          // U rows (j) per conv block
#define NB     (N_IN / TM)  // 512 conv blocks
#define STRIP  768          // 767 used, padded
#define E0B    512          // encoder-L0 blocks
#define E0R    (N_IN / E0B) // 256 rows per block

// ---------------- workspace layout (float offsets) ----------------
#define OFF_PART  0                       // E0B*512 floats (enc0 partials)
#define OFF_H0    (E0B * L0)
#define OFF_H1    (OFF_H0 + L0)
#define OFF_Z     (OFF_H1 + L1)
#define OFF_WIN   (OFF_Z + NLAT)          // 32768 bf16 (as ushort) = 16384 floats
#define OFF_STRIP (OFF_WIN + 16384)       // NB * STRIP floats

typedef __attribute__((ext_vector_type(16))) __bf16 v16bf;
typedef __attribute__((ext_vector_type(8)))  float  v8f;
typedef __attribute__((ext_vector_type(4)))  float  v4f;
typedef __attribute__((ext_vector_type(4)))  unsigned int tdm_v4u;
typedef __attribute__((ext_vector_type(8)))  int          tdm_v8i;
typedef __attribute__((ext_vector_type(4)))  int          tdm_v4i;

#if defined(__has_builtin)
#if __has_builtin(__builtin_amdgcn_tensor_load_to_lds) && \
    __has_builtin(__builtin_amdgcn_s_wait_tensorcnt)
#define USE_TDM 1
#endif
#endif

union Frag16 { uint4 q[2]; v16bf v; };

__device__ __forceinline__ float sigmoidf_(float x) { return 1.0f / (1.0f + __expf(-x)); }
__device__ __forceinline__ float swishf_(float x)   { return x * sigmoidf_(x); }
__device__ __forceinline__ unsigned short f2bf(float f) {   // round-to-nearest-even
  unsigned u = __float_as_uint(f);
  u += 0x7FFFu + ((u >> 16) & 1u);
  return (unsigned short)(u >> 16);
}

// ---- encoder layer 0: 512 blocks x 256 rows; b128 NT streams; deterministic ----
__global__ void k_enc0_partial(const float* __restrict__ x,
                               const float* __restrict__ W0,
                               float* __restrict__ part) {
  __shared__ float red[L0];
  const int b = blockIdx.x, t = threadIdx.x;
  const int c = (t & 127) * 4;   // column group (float4)
  const int half = t >> 7;       // j parity
  const int j0 = b * E0R;
  v4f acc = {0.f, 0.f, 0.f, 0.f};
  const float* base = W0 + (size_t)j0 * L0 + c;
  for (int j = half; j < E0R; j += 2) {
    float xv = x[j0 + j];
    v4f wv = __builtin_nontemporal_load((const v4f*)(base + (size_t)j * L0));
    acc += wv * xv;
  }
  if (half) *(v4f*)(red + c) = acc;
  __syncthreads();
  if (!half) {
    acc += *(const v4f*)(red + c);
    *(v4f*)(part + (size_t)b * L0 + c) = acc;
  }
}

__global__ void k_enc0_reduce(const float* __restrict__ part,
                              const float* __restrict__ b0,
                              float* __restrict__ h0) {
  int c = blockIdx.x * 256 + threadIdx.x;  // 0..511
  float s = 0.f;
  for (int b = 0; b < E0B; ++b) s += part[(size_t)b * L0 + c];
  h0[c] = swishf_(s + b0[c]);
}

// ---- encoder layer 1: 512 -> 256, one block ----
__global__ void k_enc1(const float* __restrict__ h0, const float* __restrict__ W1,
                       const float* __restrict__ b1, float* __restrict__ h1) {
  __shared__ float sh[L0];
  int t = threadIdx.x;
  sh[t] = h0[t]; sh[t + 256] = h0[t + 256];
  __syncthreads();
  float acc = 0.f;
  for (int j = 0; j < L0; ++j) acc += sh[j] * W1[(size_t)j * L1 + t];
  h1[t] = swishf_(acc + b1[t]);
}

// ---- z, smoothness map, and bf16 window table winT[t*64+i] ----
__global__ void k_latent(const float* __restrict__ h1g,
                         const float* __restrict__ W2, const float* __restrict__ b2,
                         const float* __restrict__ mW0, const float* __restrict__ mb0,
                         const float* __restrict__ mW1, const float* __restrict__ mb1,
                         const float* __restrict__ mW2, const float* __restrict__ mb2,
                         float* __restrict__ zg, unsigned short* __restrict__ winT) {
  __shared__ float sh1[L1], sz[NLAT], s0[NLAT], s1[NLAT], sw[NLAT], lps[NLAT * 4];
  int t = threadIdx.x;
  if (t < L1) sh1[t] = h1g[t];
  __syncthreads();
  if (t < NLAT) {
    float a = 0.f;
    for (int j = 0; j < L1; ++j) a += sh1[j] * W2[j * NLAT + t];
    float z = a + b2[t];
    sz[t] = z; zg[t] = z;
  }
  __syncthreads();
  if (t < NLAT) {
    float a = 0.f;
    for (int j = 0; j < NLAT; ++j) a += sz[j] * mW0[j * NLAT + t];
    s0[t] = sigmoidf_(a + mb0[t]);
  }
  __syncthreads();
  if (t < NLAT) {
    float a = 0.f;
    for (int j = 0; j < NLAT; ++j) a += s0[j] * mW1[j * NLAT + t];
    s1[t] = sigmoidf_(a + mb1[t]);
  }
  __syncthreads();
  if (t < NLAT) {
    float a = 0.f;
    for (int j = 0; j < NLAT; ++j) a += s1[j] * mW2[j * NLAT + t];
    sw[t] = sigmoidf_(a + mb2[t]);
  }
  __syncthreads();
  // bubble window: relu(1 - (k-256)^2 / (w*256)^2), normalized; 4 threads per latent
  int g = t >> 2, r = t & 3;
  float inv = 1.0f / (sw[g] * 256.0f);
  float ps = 0.f;
  for (int k = r; k < WIN; k += 4) {
    float d = ((float)k - 256.0f) * inv;
    float v = 1.0f - d * d;
    ps += v > 0.f ? v : 0.f;
  }
  lps[t] = ps;
  __syncthreads();
  float rinv = 1.0f / (lps[g*4] + lps[g*4+1] + lps[g*4+2] + lps[g*4+3]);
  for (int k = r; k < WIN; k += 4) {
    float d = ((float)k - 256.0f) * inv;
    float v = 1.0f - d * d;
    v = v > 0.f ? v : 0.f;
    winT[(size_t)k * NLAT + g] = f2bf(v * rinv);   // [t][i] layout for B frags
  }
}

// ---- decoder+conv as GEMM: G = U(256x64) @ Win(64x512) via bf16 WMMA,
//      TDM-staged window tile with hardware pad, anti-diagonal LDS reduce ----
__global__ void __launch_bounds__(256)
k_conv_wmma(const float* __restrict__ zg,
            const float* __restrict__ decW,
            const float* __restrict__ decb,
            const unsigned short* __restrict__ winT,
            float* __restrict__ strips) {
  extern __shared__ char smem[];
  unsigned short* sW = (unsigned short*)smem;       // [512][72] bf16 (padded rows)
  unsigned short* sU = sW + (size_t)WIN * 72;       // [256][72] bf16 (padded rows)
  float* sStrip = (float*)(sU + (size_t)TM * 72);   // [768]
  float* sZ = sStrip + STRIP;                       // [64]

  const int tid = threadIdx.x;
  const int b   = blockIdx.x;
  const int j0  = b * TM;

  if (tid < NLAT) sZ[tid] = zg[tid];

#ifdef USE_TDM
  if (tid < 32) {  // wave 0 issues one TDM DMA: 512x64 bf16 tile, 16B pad per row
    unsigned lds = (unsigned)(size_t)(void*)sW;               // LDS byte offset
    unsigned long long ga = (unsigned long long)(size_t)winT; // 57-bit global addr
    tdm_v4u g0;
    g0[0] = 1u;                                       // count=1 (valid descriptor)
    g0[1] = lds;                                      // lds_addr
    g0[2] = (unsigned)(ga & 0xFFFFFFFFu);             // global_addr[31:0]
    g0[3] = (unsigned)((ga >> 32) & 0x01FFFFFFu) | (2u << 30); // addr[56:32]|type=2
    tdm_v8i g1;
    g1[0] = (1 << 16)     // data_size = 2 bytes
          | (1 << 20)     // pad_enable
          | (4 << 22)     // pad_interval: 32 DWORDs (one 128B row)
          | (3 << 25);    // pad_amount: 4 DWORDs (16B) -> 144B LDS row stride
    g1[1] = (NLAT << 16); // tensor_dim0 = 64 (low16)
    g1[2] = (WIN << 16);  // tensor_dim0 hi=0 | tensor_dim1 = 512 (low16)
    g1[3] = (NLAT << 16); // tensor_dim1 hi=0 | tile_dim0 = 64
    g1[4] = WIN;          // tile_dim1 = 512, tile_dim2 = 0
    g1[5] = NLAT;         // tensor_dim0_stride = 64 (low32)
    g1[6] = 0;            // stride0 hi | tensor_dim1_stride (unused, 2D)
    g1[7] = 0;
    tdm_v4i zz = {0, 0, 0, 0};
#if __clang_major__ >= 23
    tdm_v8i zz8 = {0, 0, 0, 0, 0, 0, 0, 0};
    __builtin_amdgcn_tensor_load_to_lds(g0, g1, zz, zz, zz8, 0);
#else
    __builtin_amdgcn_tensor_load_to_lds(g0, g1, zz, zz, 0);
#endif
  }
#else
  {
    const unsigned* wsrc = (const unsigned*)winT;
    for (int d = tid; d < WIN * 32; d += 256) {
      int tt = d >> 5, p = d & 31;
      ((unsigned*)(sW + (size_t)tt * 72))[p] = wsrc[d];
    }
  }
#endif

  for (int s = tid; s < STRIP; s += 256) sStrip[s] = 0.f;
  __syncthreads();  // sZ visible; TDM still in flight, overlapped with U staging

  // stage U tile: u_i[j] = z_i * dec_W[i][j] + dec_b[i][j]; single-use -> NT loads
  for (int i = 0; i < NLAT; ++i) {
    size_t gidx = (size_t)i * N_IN + j0 + tid;
    float u = sZ[i] * __builtin_nontemporal_load(decW + gidx)
                    + __builtin_nontemporal_load(decb + gidx);
    sU[(size_t)tid * 72 + i] = f2bf(u);
  }
#ifdef USE_TDM
  if (tid < 32) __builtin_amdgcn_s_wait_tensorcnt(0);  // wave 0: DMA complete
#endif
  __syncthreads();

  const int w    = tid >> 5;          // 8 waves, wave w owns M-tiles {2w, 2w+1}
  const int lane = tid & 31;
  const int am   = lane & 15;
  const int ah   = (lane >> 4) << 3;  // A half-wave K interleave (+8)
  const int bh   = (lane >> 4) << 4;  // B half-wave K offset (+16)

  // A fragments (16-bit A 16x32 layout, per ISA table): 2 mtiles x 2 k-steps
  Frag16 a[2][2];
#pragma unroll
  for (int mtl = 0; mtl < 2; ++mtl) {
    const unsigned short* ur = sU + (size_t)((2 * w + mtl) * 16 + am) * 72;
    a[mtl][0].q[0] = *(const uint4*)(ur + 0  + ah);
    a[mtl][0].q[1] = *(const uint4*)(ur + 16 + ah);
    a[mtl][1].q[0] = *(const uint4*)(ur + 32 + ah);
    a[mtl][1].q[1] = *(const uint4*)(ur + 48 + ah);
  }

  for (int n = 0; n < 32; ++n) {      // 32 N-tiles cover t = 0..511
    Frag16 b0f, b1f;                  // K-steps 0 and 1 of Win(64 x 16)
    const unsigned short* wr = sW + (size_t)(n * 16 + am) * 72;
    b0f.q[0] = *(const uint4*)(wr + 0  + ah);
    b0f.q[1] = *(const uint4*)(wr + 16 + ah);
    b1f.q[0] = *(const uint4*)(wr + 32 + ah);
    b1f.q[1] = *(const uint4*)(wr + 48 + ah);
#pragma unroll
    for (int mtl = 0; mtl < 2; ++mtl) {
      v8f c = {0.f, 0.f, 0.f, 0.f, 0.f, 0.f, 0.f, 0.f};
      c = __builtin_amdgcn_wmma_f32_16x16x32_bf16(false, a[mtl][0].v, false, b0f.v,
                                                  (short)0, c, false, false);
      c = __builtin_amdgcn_wmma_f32_16x16x32_bf16(false, a[mtl][1].v, false, b1f.v,
                                                  (short)0, c, false, false);
      // D lane map: VGPR r -> M = r + 8*(lane>=16), N = lane&15 ; strip idx = m + t
      int baseS = (2 * w + mtl) * 16 + ah + n * 16 + (lane & 15);
#pragma unroll
      for (int r = 0; r < 8; ++r) atomicAdd(&sStrip[baseS + r], c[r]);
    }
  }
  __syncthreads();
  for (int s = tid; s < STRIP; s += 256)
    strips[(size_t)b * STRIP + s] = sStrip[s];
}

// ---- deterministic gather of overlapping strips: out[k] = sum_b strip[b][k-256b+255] ----
__global__ void k_gather(const float* __restrict__ strips, float* __restrict__ out) {
  int k = blockIdx.x * 256 + threadIdx.x;
  int num = k - 511;
  int bmin = (num <= 0) ? 0 : ((num + 255) >> 8);
  int bmax = (k + 255) >> 8;
  if (bmax > NB - 1) bmax = NB - 1;
  float acc = 0.f;
  for (int bb = bmin; bb <= bmax; ++bb)
    acc += strips[(size_t)bb * STRIP + (k - bb * TM + 255)];
  out[k] = acc;
}

extern "C" void kernel_launch(void* const* d_in, const int* in_sizes, int n_in,
                              void* d_out, int out_size, void* d_ws, size_t ws_size,
                              hipStream_t stream) {
  const float* x   = (const float*)d_in[0];
  const float* eW0 = (const float*)d_in[1];
  const float* eb0 = (const float*)d_in[2];
  const float* eW1 = (const float*)d_in[3];
  const float* eb1 = (const float*)d_in[4];
  const float* eW2 = (const float*)d_in[5];
  const float* eb2 = (const float*)d_in[6];
  const float* mW0 = (const float*)d_in[7];
  const float* mb0 = (const float*)d_in[8];
  const float* mW1 = (const float*)d_in[9];
  const float* mb1 = (const float*)d_in[10];
  const float* mW2 = (const float*)d_in[11];
  const float* mb2 = (const float*)d_in[12];
  const float* dW  = (const float*)d_in[13];
  const float* db  = (const float*)d_in[14];

  float* out = (float*)d_out;
  float* wsf = (float*)d_ws;
  float* part = wsf + OFF_PART;
  float* h0   = wsf + OFF_H0;
  float* h1   = wsf + OFF_H1;
  float* z    = wsf + OFF_Z;
  unsigned short* winT = (unsigned short*)(wsf + OFF_WIN);
  float* strips = wsf + OFF_STRIP;

  k_enc0_partial<<<E0B, 256, 0, stream>>>(x, eW0, part);
  k_enc0_reduce<<<2, 256, 0, stream>>>(part, eb0, h0);
  k_enc1<<<1, 256, 0, stream>>>(h0, eW1, eb1, h1);
  k_latent<<<1, 256, 0, stream>>>(h1, eW2, eb2, mW0, mb0, mW1, mb1, mW2, mb2, z, winT);

  size_t smem = (size_t)WIN * 72 * 2 + (size_t)TM * 72 * 2 + STRIP * 4 + NLAT * 4;
  k_conv_wmma<<<NB, 256, smem, stream>>>(z, dW, db, winT, strips);
  k_gather<<<N_IN / 256, 256, 0, stream>>>(strips, out);
}
